// RNN_60730837565520
// MI455X (gfx1250) — matrix operand order repaired and checked by reading, not compiled, not run on hardware
//
#include <hip/hip_runtime.h>

// RNN block for MI455X (gfx1250).
//   pre:     input -> bf16; w_in/w_out -> bf16, chunk-tiled K-transposed layout
//   stage 1: x = input @ w_in + b_in   (async-DMA double-buffered WMMA GEMM)
//   stage 2: s_t = tanh(s_{t-1} @ W_n + x_t) per head; W_n via TDM, register
//            B-frags; x_t double-buffered via global_load_async_to_lds
//   stage 3: out = y @ w_out           (same GEMM; A already bf16)
// Workspace: [x f32 128MB | y bf16 64MB | input bf16 64MB | w_in^T 8MB | w_out^T 8MB]

#define BATCH  4
#define SEQ    4096
#define DMODEL 2048
#define NHEADS 16
#define HDIM   128

typedef __attribute__((ext_vector_type(16))) __bf16       v16bf;
typedef __attribute__((ext_vector_type(8)))  float        v8f;
typedef __attribute__((ext_vector_type(4)))  unsigned int v4u;
typedef __attribute__((ext_vector_type(8)))  unsigned int v8u;

union Frag {
    uint4 u[2];
    v16bf v;
};

__device__ __forceinline__ unsigned int pack2bf(float a, float b) {
    __bf16 lo = (__bf16)a;
    __bf16 hi = (__bf16)b;
    unsigned short ulo = __builtin_bit_cast(unsigned short, lo);
    unsigned short uhi = __builtin_bit_cast(unsigned short, hi);
    return (unsigned int)ulo | ((unsigned int)uhi << 16);
}

// Flat shared-memory pointer -> LDS byte offset (low 32 bits of the flat address).
__device__ __forceinline__ unsigned lds_off(const void* p) {
    return (unsigned)(size_t)p;
}

// Async global->LDS copy, GVS mode: saddr(u64 base) + vaddr(i32 offset), 16B/lane.
__device__ __forceinline__ void async_load_b128(unsigned ldsAddr, unsigned voff,
                                                const void* base) {
    asm volatile("global_load_async_to_lds_b128 %0, %1, %2"
                 :: "v"(ldsAddr), "v"(voff), "s"(base)
                 : "memory");
}

// TDM: 2D f32 tile (rows x 128) from global tensor (stride 128) into LDS.
__device__ __forceinline__ void tdm_load_2d(unsigned ldsAddr, const void* gaddr,
                                            unsigned rows) {
    unsigned long long ga = (unsigned long long)(size_t)gaddr;
    v4u g0 = { 1u, ldsAddr, (unsigned)ga,
               ((unsigned)(ga >> 32) & 0x01FFFFFFu) | 0x80000000u };
    v8u g1 = { 0x20000u,        // [17:16] data_size = 2 -> 4 bytes
               128u << 16,      // [63:48]  tensor_dim0 = 128
               128u << 16,      // [95:80]  tensor_dim1 = 128
               128u << 16,      // [127:112] tile_dim0 = 128
               rows,            // [143:128] tile_dim1
               128u,            // [207:160] tensor_dim0_stride = 128
               0u, 0u };
    v4u g2 = { 0u, 0u, 0u, 0u };
    v4u g3 = { 0u, 0u, 0u, 0u };
    asm volatile("tensor_load_to_lds %0, %1, %2, %3"
                 :: "s"(g0), "s"(g1), "s"(g2), "s"(g3)
                 : "memory");
}

// branch-free tanh: 1 - 2/(exp(2x)+1)
__device__ __forceinline__ float fast_tanh(float x) {
    float e = __expf(2.0f * x);
    return 1.0f - 2.0f * __builtin_amdgcn_rcpf(e + 1.0f);
}

// ---------------------------------------------------------------------------
// Pre-pass: plain f32 -> bf16 cast (4 elements / thread).
// ---------------------------------------------------------------------------
__global__ __launch_bounds__(256) void cast_bf16_kernel(
    const float* __restrict__ src, unsigned short* __restrict__ dst) {
    size_t i = ((size_t)blockIdx.x * 256 + threadIdx.x) * 4;
    float4 v = *(const float4*)(src + i);
    *(unsigned int*)(dst + i)     = pack2bf(v.x, v.y);
    *(unsigned int*)(dst + i + 2) = pack2bf(v.z, v.w);
}

// ---------------------------------------------------------------------------
// Pre-pass: W[k][e] f32 -> W2 bf16 chunk-tiled: W2[(kc*2048 + e)*32 + kk],
// kc = k/32, kk = k%32.  A block's 128x32 B-tile becomes one 8KB slab.
// ---------------------------------------------------------------------------
__global__ __launch_bounds__(256) void wtile_bf16_kernel(
    const float* __restrict__ W, unsigned short* __restrict__ W2) {
    size_t idx = (size_t)blockIdx.x * 256 + threadIdx.x;  // over 2048*2048
    int k = (int)(idx >> 11);
    int e = (int)(idx & 2047);
    int kc = k >> 5, kk = k & 31;
    __bf16 b = (__bf16)W[idx];
    W2[((size_t)kc * DMODEL + e) * 32 + kk] = __builtin_bit_cast(unsigned short, b);
}

// ---------------------------------------------------------------------------
// Async-DMA double-buffered WMMA GEMM:
//   C[M x 2048] = A[M x 2048](bf16) * W2(bf16, tiled) (+ bias)
// 256 threads (8 waves), 128x128 C tile, K-chunks of 32, LDS double buffer.
// Inner loop: 4 async b128 issues/wave + s_wait_asynccnt + 18 ds_load + 8 wmma.
// ---------------------------------------------------------------------------
template <bool HAS_BIAS>
__global__ __launch_bounds__(256) void wmma_gemm_kernel(
    const unsigned short* __restrict__ Abf, const unsigned short* __restrict__ W2,
    const float* __restrict__ bias, float* __restrict__ C) {
    constexpr int K  = DMODEL;
    constexpr int Nn = DMODEL;
    constexpr int NKC = K / 32;
    __shared__ __bf16 As[2][128][40];  // 80B pitch, 16B-aligned, conflict-free
    __shared__ __bf16 Bs[2][128][40];

    const int tid     = threadIdx.x;
    const int lane    = tid & 31;
    const int wid     = tid >> 5;
    const int halfSel = lane >> 4;
    const int lrow    = lane & 15;
    const int n0      = blockIdx.x * 128;
    const int m0      = blockIdx.y * 128;
    const unsigned asBase = lds_off(&As[0][0][0]);
    const unsigned bsBase = lds_off(&Bs[0][0][0]);

    v8f acc[8] = {};

    // issue chunk 0 into buffer 0
    auto issue_chunk = [&](int kc, int buf) {
#pragma unroll
        for (int i = 0; i < 2; ++i) {  // A tile: 128 rows x 64B
            int linear = tid + i * 256;
            int row = linear >> 2, q = linear & 3;
            unsigned voff = (unsigned)((((size_t)(m0 + row) * K) + kc * 32 + q * 8) * 2);
            async_load_b128(asBase + (unsigned)(buf * 10240 + row * 80 + q * 16), voff, Abf);
        }
#pragma unroll
        for (int i = 0; i < 2; ++i) {  // B tile: one contiguous 8KB slab in W2
            int linear = tid + i * 256;
            int e = linear >> 2, q = linear & 3;
            unsigned voff = (unsigned)((((size_t)kc * DMODEL + n0 + e) * 32 + q * 8) * 2);
            async_load_b128(bsBase + (unsigned)(buf * 10240 + e * 80 + q * 16), voff, W2);
        }
    };
    issue_chunk(0, 0);

    for (int kc = 0; kc < NKC; ++kc) {
        const int buf = kc & 1;
        if (kc + 1 < NKC) {
            issue_chunk(kc + 1, buf ^ 1);  // prev iter's tail barrier made buf^1 free
            // ASYNCcnt in-order per wave: <=4 outstanding => chunk kc landed
            asm volatile("s_wait_asynccnt 0x4" ::: "memory");
        } else {
            asm volatile("s_wait_asynccnt 0x0" ::: "memory");
        }
        __syncthreads();  // chunk kc visible to all waves

        Frag a;  // A frag (ISA 16-bit 16x32 layout)
        {
            int row = wid * 16 + lrow;
            int kb  = halfSel * 8;
            a.u[0] = *(const uint4*)&As[buf][row][kb];
            a.u[1] = *(const uint4*)&As[buf][row][kb + 16];
        }
#pragma unroll
        for (int j = 0; j < 8; ++j) {
            Frag b;  // B 32x16: lanes 0-15 K=0..7,16..23, lanes 16-31 K=8..15,24..31
            int e  = j * 16 + lrow;
            int kb = halfSel * 8;
            b.u[0] = *(const uint4*)&Bs[buf][e][kb];
            b.u[1] = *(const uint4*)&Bs[buf][e][kb + 16];
            acc[j] = __builtin_amdgcn_wmma_f32_16x16x32_bf16(
                false, a.v, false, b.v, (short)0, acc[j], false, false);
        }
        __syncthreads();  // frag reads of buf done before it is overwritten
    }

    // epilogue: D layout VGPR r -> M = r + 8*halfSel, N = lane%16
#pragma unroll
    for (int j = 0; j < 8; ++j) {
        int   col = n0 + j * 16 + lrow;
        float bv  = 0.0f;
        if constexpr (HAS_BIAS) bv = bias[col];
#pragma unroll
        for (int r = 0; r < 8; ++r) {
            int m = m0 + wid * 16 + r + halfSel * 8;
            C[(size_t)m * Nn + col] = acc[j][r] + bv;
        }
    }
}

// ---------------------------------------------------------------------------
// Sequential scan: one block per head. W_n staged via TDM, converted to
// register-resident WMMA B-frags. x_t double-buffered via async global->LDS.
// ---------------------------------------------------------------------------
__global__ __launch_bounds__(256) void wmma_scan_kernel(
    const float* __restrict__ x,       // (B,S,D) f32
    const float* __restrict__ state0,  // (B,N,H) f32
    const float* __restrict__ sw,      // (N,H,H) f32
    unsigned short* __restrict__ y) {  // (B,S,D) bf16 out
    __shared__ __bf16 Ws[128][136];        // W_n [out_col][h], 272B pitch
    __shared__ __bf16 Ss[16][136];         // state A-tile, rows >= BATCH zero
    __shared__ float  Wstage[32][128];     // TDM staging (16KB)
    __shared__ float  xb[2][BATCH][HDIM];  // x_t double buffer (4KB)

    const int tid     = threadIdx.x;
    const int lane    = tid & 31;
    const int wid     = tid >> 5;
    const int n       = blockIdx.x;
    const int halfSel = lane >> 4;
    const int lrow    = lane & 15;

    for (int i = tid; i < 16 * 136; i += 256) ((__bf16*)Ss)[i] = (__bf16)0.0f;
    __syncthreads();
    for (int i = tid; i < BATCH * HDIM; i += 256) {
        int b = i >> 7;
        int h = i & 127;
        Ss[b][h] = (__bf16)state0[((size_t)b * NHEADS + n) * HDIM + h];
    }

    // W_n via TDM in 4 chunks of 32 rows; convert f32 -> bf16 K-transposed
    const float*   wn       = sw + (size_t)n * HDIM * HDIM;
    const unsigned stageOff = lds_off(&Wstage[0][0]);
    for (int c = 0; c < 4; ++c) {
        if (wid == 0) {
            tdm_load_2d(stageOff, wn + (size_t)c * 32 * HDIM, 32u);
            __builtin_amdgcn_s_wait_tensorcnt((short)0);
        }
        __syncthreads();
        for (int i = tid; i < 32 * HDIM; i += 256) {
            int r = i >> 7;
            int e = i & 127;
            Ws[e][c * 32 + r] = (__bf16)Wstage[r][e];
        }
        __syncthreads();
    }

    Frag bfrag[4];
#pragma unroll
    for (int c = 0; c < 4; ++c) {
        int e  = wid * 16 + lrow;
        int kb = c * 32 + halfSel * 8;
        bfrag[c].u[0] = *(const uint4*)&Ws[e][kb];
        bfrag[c].u[1] = *(const uint4*)&Ws[e][kb + 16];
    }

    const int      col   = wid * 16 + lrow;
    const unsigned xOff0 = lds_off(&xb[0][0][0]);
    const unsigned xOff1 = lds_off(&xb[1][0][0]);

    if (tid < 128) {  // async x_t load for step 0 (2KB slab, 16B/lane)
        int b = tid >> 5, cc = tid & 31;
        unsigned voff =
            (unsigned)((((size_t)b * SEQ + 0) * DMODEL + (size_t)n * HDIM) * 4 + cc * 16);
        async_load_b128(xOff0 + (unsigned)((b * HDIM + cc * 4) * 4), voff, x);
    }

    for (int s = 0; s < SEQ; ++s) {
        if (s + 1 < SEQ && tid < 128) {
            int b = tid >> 5, cc = tid & 31;
            unsigned dst = ((s + 1) & 1) ? xOff1 : xOff0;
            unsigned voff = (unsigned)((((size_t)b * SEQ + (s + 1)) * DMODEL +
                                        (size_t)n * HDIM) * 4 + cc * 16);
            async_load_b128(dst + (unsigned)((b * HDIM + cc * 4) * 4), voff, x);
        }

        Frag a0, a1, a2, a3;
        {
            int kb = halfSel * 8;
            a0.u[0] = *(const uint4*)&Ss[lrow][kb];
            a0.u[1] = *(const uint4*)&Ss[lrow][kb + 16];
            a1.u[0] = *(const uint4*)&Ss[lrow][32 + kb];
            a1.u[1] = *(const uint4*)&Ss[lrow][32 + kb + 16];
            a2.u[0] = *(const uint4*)&Ss[lrow][64 + kb];
            a2.u[1] = *(const uint4*)&Ss[lrow][64 + kb + 16];
            a3.u[0] = *(const uint4*)&Ss[lrow][96 + kb];
            a3.u[1] = *(const uint4*)&Ss[lrow][96 + kb + 16];
        }
        v8f acc0 = {}, acc1 = {};
        acc0 = __builtin_amdgcn_wmma_f32_16x16x32_bf16(false, a0.v, false, bfrag[0].v,
                                                       (short)0, acc0, false, false);
        acc1 = __builtin_amdgcn_wmma_f32_16x16x32_bf16(false, a1.v, false, bfrag[1].v,
                                                       (short)0, acc1, false, false);
        acc0 = __builtin_amdgcn_wmma_f32_16x16x32_bf16(false, a2.v, false, bfrag[2].v,
                                                       (short)0, acc0, false, false);
        acc1 = __builtin_amdgcn_wmma_f32_16x16x32_bf16(false, a3.v, false, bfrag[3].v,
                                                       (short)0, acc1, false, false);
        v8f acc = acc0 + acc1;

        if (s + 1 < SEQ) {
            asm volatile("s_wait_asynccnt 0x1" ::: "memory");
        } else {
            asm volatile("s_wait_asynccnt 0x0" ::: "memory");
        }
        __syncthreads();  // Ss reads done + x buffer visible

        if (halfSel == 0) {
            const float* xs = &xb[s & 1][0][0];
#pragma unroll
            for (int r = 0; r < BATCH; ++r) {
                float  v  = fast_tanh(acc[r] + xs[r * HDIM + col]);
                __bf16 bv = (__bf16)v;
                y[((size_t)r * SEQ + s) * DMODEL + (size_t)n * HDIM + col] =
                    __builtin_bit_cast(unsigned short, bv);
                Ss[r][col] = bv;
            }
        }
        __syncthreads();
    }
}

extern "C" void kernel_launch(void* const* d_in, const int* in_sizes, int n_in,
                              void* d_out, int out_size, void* d_ws, size_t ws_size,
                              hipStream_t stream) {
    (void)in_sizes; (void)n_in; (void)out_size; (void)ws_size;
    const float* input  = (const float*)d_in[0];
    const float* state0 = (const float*)d_in[1];
    const float* w_in   = (const float*)d_in[2];
    const float* b_in   = (const float*)d_in[3];
    const float* sw     = (const float*)d_in[4];
    const float* w_out  = (const float*)d_in[5];
    float*       out    = (float*)d_out;

    const size_t MD = (size_t)BATCH * SEQ * DMODEL;  // 33.5M
    const size_t WD = (size_t)DMODEL * DMODEL;       // 4.2M
    char* ws = (char*)d_ws;
    float*          xbuf   = (float*)ws;                              // 128MB
    unsigned short* ybuf   = (unsigned short*)(ws + 4 * MD);          // 64MB
    unsigned short* in_bf  = (unsigned short*)(ws + 6 * MD);          // 64MB
    unsigned short* winT   = (unsigned short*)(ws + 8 * MD);          // 8MB
    unsigned short* woutT  = (unsigned short*)(ws + 8 * MD + 2 * WD); // 8MB

    // pre-pass conversions (one-time, bandwidth-trivial)
    cast_bf16_kernel<<<(unsigned)(MD / 4 / 256), 256, 0, stream>>>(input, in_bf);
    wtile_bf16_kernel<<<(unsigned)(WD / 256), 256, 0, stream>>>(w_in, winT);
    wtile_bf16_kernel<<<(unsigned)(WD / 256), 256, 0, stream>>>(w_out, woutT);

    dim3 gblk(DMODEL / 128, (BATCH * SEQ) / 128);  // 16 x 128 blocks
    wmma_gemm_kernel<true><<<gblk, 256, 0, stream>>>(in_bf, winT, b_in, xbuf);
    wmma_scan_kernel<<<NHEADS, 256, 0, stream>>>(xbuf, state0, sw, ybuf);
    wmma_gemm_kernel<false><<<gblk, 256, 0, stream>>>(ybuf, woutT, nullptr, out);
}
